// DeepBidirectionalLSTM_31525059953172
// MI455X (gfx1250) — matrix operand order repaired and checked by reading, not compiled, not run on hardware
//
#include <hip/hip_runtime.h>

// ---------------- problem constants (from reference) ----------------
#define B_    64
#define T_    512
#define E_    256
#define U_    512
#define G4U_  2048      // 4*U gate columns
#define KTOT_ 768       // E + U
#define AXSTR_ 264      // x-tile LDS row stride in halfs (256 + 8 pad; 528B, 16B mult)
#define AHSTR_ 520      // h-tile LDS row stride in halfs (512 + 8 pad; 1040B, 16B mult)
#define NWG_  16        // workgroups per direction

typedef __attribute__((ext_vector_type(16))) _Float16 v16h;
typedef __attribute__((ext_vector_type(8)))  _Float16 v8h;
typedef __attribute__((ext_vector_type(4)))  _Float16 v4h;
typedef __attribute__((ext_vector_type(8)))  float    v8f;
typedef __attribute__((ext_vector_type(4)))  unsigned int v4u;
typedef __attribute__((ext_vector_type(8)))  int      v8i_;
typedef __attribute__((ext_vector_type(4)))  int      v4i_;

// Branch-free fast sigmoid: v_exp_f32 + v_rcp_f32, saturates at 0/1.
__device__ __forceinline__ float sigmoid_fast(float x) {
  return __builtin_amdgcn_rcpf(1.0f + __expf(-x));
}
// Branch-free fast tanh: e = exp(-2|x|) in (0,1] -> no overflow path needed.
__device__ __forceinline__ float tanh_fast(float x) {
  float e = __expf(-2.0f * __builtin_fabsf(x));
  float t = (1.0f - e) * __builtin_amdgcn_rcpf(1.0f + e);
  return __builtin_copysignf(t, x);
}

#if __has_builtin(__builtin_amdgcn_tensor_load_to_lds)
#define HAVE_TDM 1
// TDM 2D tile load Global->LDS with per-row LDS padding (D# per ISA 8.3/8.4).
// rowHalfs: tile width in 2-byte elements; rows: tile height.
// padInterval: pad after 2^(padInterval+3) bytes; padAmount: (padAmount+1) DWORDs.
// 6-arg builtin form (clang-23 / therock): (u32x4, i32x8, i32x4, i32x4, i32x8, i32).
__device__ __forceinline__ void tdm_load_2d(unsigned ldsAddr, const void* gsrc,
                                            unsigned rowHalfs, unsigned rows,
                                            unsigned padInterval, unsigned padAmount) {
  unsigned long long ga = (unsigned long long)(uintptr_t)gsrc;
  v4u g0;
  g0[0] = 1u;                                        // count=1 (valid), flags=0
  g0[1] = ldsAddr;                                   // LDS byte address
  g0[2] = (unsigned)(ga & 0xFFFFFFFFu);              // global_addr[31:0]
  g0[3] = (unsigned)((ga >> 32) & 0x01FFFFFFu)       // global_addr[56:32]
        | (2u << 30);                                // type=2 ("image")
  v8i_ g1;
  g1[0] = (int)((1u << 16)                           // data_size=1 -> 2 bytes
              | (1u << 20)                           // pad_enable
              | (padInterval << 22)
              | (padAmount << 25));                  // workgroup_mask=0 (no cluster)
  g1[1] = (int)((rowHalfs & 0xFFFFu) << 16);         // tensor_dim0[15:0] @63:48
  g1[2] = (int)((rows & 0xFFFFu) << 16);             // dim0[31:16]=0 | tensor_dim1 lo
  g1[3] = (int)((rowHalfs & 0xFFFFu) << 16);         // dim1 hi=0 | tile_dim0
  g1[4] = (int)(rows & 0xFFFFu);                     // tile_dim1 | tile_dim2=0
  g1[5] = (int)rowHalfs;                             // tensor_dim0_stride[31:0]
  g1[6] = 0;                                         // stride hi | dim1_stride lo
  g1[7] = 0;
  v4i_ zz4 = {0, 0, 0, 0};                           // groups 2/3 unused (2D tensor)
  v8i_ zz8 = {0, 0, 0, 0, 0, 0, 0, 0};
  __builtin_amdgcn_tensor_load_to_lds(g0, g1, zz4, zz4, zz8, 0);
}
#else
#define HAVE_TDM 0
#endif

// ---------------- embedding gather: fp32 emb rows -> fp16 X[T][B][E] --------
__global__ void embed_gather_f16(const int* __restrict__ sent,
                                 const float* __restrict__ emb,
                                 _Float16* __restrict__ X) {
  int idx = blockIdx.x * blockDim.x + threadIdx.x;   // one float4 -> v4h
  const int total = T_ * B_ * (E_ / 4);
  if (idx >= total) return;
  int t   = idx >> 12;           // / (B_*E_/4) == 4096
  int rem = idx & 4095;
  int b   = rem >> 6;            // / (E_/4)
  int e4  = rem & 63;
  int token = sent[b * T_ + t];
  float4 v = ((const float4*)emb)[(size_t)token * (E_ / 4) + e4];
  v4h o = { (_Float16)v.x, (_Float16)v.y, (_Float16)v.z, (_Float16)v.w };
  ((v4h*)X)[idx] = o;
}

// ---------------- fp32 -> fp16 weight conversion ----------------
__global__ void convert_f16(const float* __restrict__ src,
                            _Float16* __restrict__ dst, int n4) {
  int idx = blockIdx.x * blockDim.x + threadIdx.x;
  if (idx >= n4) return;
  float4 v = ((const float4*)src)[idx];
  v4h o = { (_Float16)v.x, (_Float16)v.y, (_Float16)v.z, (_Float16)v.w };
  ((v4h*)dst)[idx] = o;
}

// ---------------- zero-init barrier + recurrent h state ----------------
__global__ void init_state(unsigned* __restrict__ bar, _Float16* __restrict__ H16) {
  int tid = blockIdx.x * blockDim.x + threadIdx.x;
  if (tid < 64) bar[tid] = 0u;
  const int nh = 2 * B_ * U_;
  for (int i = tid; i < nh; i += blockDim.x * gridDim.x) H16[i] = (_Float16)0.0f;
}

// ---------------- device-wide split barrier (per direction) ----------------
__device__ __forceinline__ void grid_barrier(unsigned* cnt, unsigned* gen, unsigned phase) {
  __syncthreads();
  if (threadIdx.x == 0) {
    __threadfence();
    unsigned prev = __hip_atomic_fetch_add(cnt, 1u, __ATOMIC_ACQ_REL,
                                           __HIP_MEMORY_SCOPE_AGENT);
    if (prev == (unsigned)NWG_ * phase - 1u) {
      __hip_atomic_store(gen, phase, __ATOMIC_RELEASE, __HIP_MEMORY_SCOPE_AGENT);
    }
    while (__hip_atomic_load(gen, __ATOMIC_ACQUIRE, __HIP_MEMORY_SCOPE_AGENT) < phase) {
      __builtin_amdgcn_s_sleep(1);
    }
    __threadfence();
  }
  __syncthreads();
}

// ---------------- persistent bidirectional LSTM recurrence ----------------
// grid = 32 WGs: blockIdx>>4 = direction, blockIdx&15 = hidden-column slice.
// Each WG: z-slice [64 x 128] (4 gates x 32 u-cols), K = 768 (x_t | h_{t-1}).
// Wave (8 per WG, wave32): Mtile = w&3 (16 batch rows), group = w>>2 (16 u-cols).
// Each wave holds all 4 gate tiles for its (16x16) patch -> cell update is
// purely lane-local; c-state stays in a v8f register for all 512 steps.
// A tile staged by the Tensor Data Mover as two padded 2D tiles (x | h).
__global__ void __launch_bounds__(256, 1)
bilstm_persistent(const _Float16* __restrict__ X,
                  const _Float16* __restrict__ Wx16f,
                  const _Float16* __restrict__ Wh16f,
                  const _Float16* __restrict__ Wx16b,
                  const _Float16* __restrict__ Wh16b,
                  const float* __restrict__ bias_f,
                  const float* __restrict__ bias_b,
                  _Float16* __restrict__ H16,   // [2][B][U] fp16 (recurrent)
                  float* __restrict__ Hout,     // [2][B][U] fp32 (final h)
                  unsigned* __restrict__ bar) {
  extern __shared__ _Float16 Alds[];            // Ax[64][264] ++ Ah[64][520] (~98 KB)
  _Float16* Ax = Alds;
  _Float16* Ah = Alds + 64 * AXSTR_;

  const int dir  = blockIdx.x >> 4;
  const int g    = blockIdx.x & 15;
  const int tid  = threadIdx.x;
  const int lane = tid & 31;
  const int wave = tid >> 5;
  const int Mtile = wave & 3;
  const int group = wave >> 2;

  const _Float16* __restrict__ Wx16 = dir ? Wx16b : Wx16f;
  const _Float16* __restrict__ Wh16 = dir ? Wh16b : Wh16f;
  const float* __restrict__ bias    = dir ? bias_b : bias_f;
  _Float16* __restrict__ Hdir = H16 + dir * B_ * U_;
  float* __restrict__ Hod     = Hout + dir * B_ * U_;
  unsigned* cnt = bar + dir * 32;
  unsigned* gen = bar + dir * 32 + 8;

  const int n      = lane & 15;              // D-layout: col within tile
  const int mBase  = Mtile * 16 + ((lane >> 4) << 3);  // D-layout row base
  const int arow   = Mtile * 16 + (lane & 15);         // A-layout: lane = row
  const int khalf8 = (lane >> 4) << 3;                 // A-layout K phase
  const int colT   = g * 32 + group * 16;    // owned u-tile base in [0,512)
  const int u      = colT + n;
  const int cb0 = 0 * U_ + colT, cb1 = 1 * U_ + colT,
            cb2 = 2 * U_ + colT, cb3 = 3 * U_ + colT;
  const float bi = bias[cb0 + n], bf = bias[cb1 + n],
              bg = bias[cb2 + n], bo = bias[cb3 + n];

  // per-lane bases inside Wx / Wh (B-layout: lane%16 = column, K phase 8*(lane/16))
  const _Float16* WxLane = Wx16 + (size_t)khalf8 * G4U_ + (lane & 15);
  const _Float16* WhLane = Wh16 + (size_t)khalf8 * G4U_ + (lane & 15);

#if HAVE_TDM
  const unsigned axOff = (unsigned)(uintptr_t)Ax;  // low 32 bits = LDS byte offset
  const unsigned ahOff = (unsigned)(uintptr_t)Ah;
#endif

  v8f cst = {0.f,0.f,0.f,0.f,0.f,0.f,0.f,0.f};
  unsigned phase = 0;

  for (int t = 0; t < T_; ++t) {
    const int tt = dir ? (T_ - 1 - t) : t;
    const _Float16* __restrict__ Xt = X + (size_t)tt * B_ * E_;

    // prefetch next step's x tile into GL2 (global_prefetch_b8)
    if (t + 1 < T_ && tid < B_) {
      const int tn = dir ? (T_ - 2 - t) : (t + 1);
      __builtin_prefetch(X + (size_t)tn * B_ * E_ + tid * E_, 0, 1);
    }

#if HAVE_TDM
    // Tensor Data Mover: stage x_t (64x256) and h_{t-1} (64x512) as padded
    // 2D tiles; one wave issues, TENSORcnt tracks, barrier publishes to WG.
    if (wave == 0) {
      tdm_load_2d(axOff, Xt,   E_, B_, /*pad: 512B rows*/ 6, /*16B*/ 3);
      tdm_load_2d(ahOff, Hdir, U_, B_, /*pad: 1024B rows*/ 7, /*16B*/ 3);
      __builtin_amdgcn_s_wait_tensorcnt(0);
    }
#else
    // manual staging fallback: 128-bit chunks through VGPRs
    {
      const uint4* xs = (const uint4*)Xt;
      const uint4* hs = (const uint4*)Hdir;
      #pragma unroll
      for (int it = 0; it < 8; ++it) {         // 64 rows x 32 uint4 (x part)
        int idx = tid + it * 256;
        int r = idx >> 5, j = idx & 31;
        *((uint4*)(Ax + r * AXSTR_) + j) = xs[r * 32 + j];
      }
      #pragma unroll
      for (int it = 0; it < 16; ++it) {        // 64 rows x 64 uint4 (h part)
        int idx = tid + it * 256;
        int r = idx >> 6, j = idx & 63;
        *((uint4*)(Ah + r * AHSTR_) + j) = hs[r * 64 + j];
      }
    }
#endif
    // all WGs must capture h(t-1) before anyone publishes h(t)
    grid_barrier(cnt, gen, ++phase);

    // bias folded into the C operand (per-column value, uniform over rows)
    v8f z0 = {bi,bi,bi,bi,bi,bi,bi,bi};
    v8f z1 = {bf,bf,bf,bf,bf,bf,bf,bf};
    v8f z2 = {bg,bg,bg,bg,bg,bg,bg,bg};
    v8f z3 = {bo,bo,bo,bo,bo,bo,bo,bo};

    // B-fragment loader: kb is uniform, 32-aligned; E_ boundary is 32-aligned,
    // so the Wx/Wh switch is wave-uniform.
    auto loadB = [&](int kb, v16h& o0, v16h& o1, v16h& o2, v16h& o3) {
      const _Float16* Wk = (kb < E_) ? (WxLane + (size_t)kb * G4U_)
                                     : (WhLane + (size_t)(kb - E_) * G4U_);
      #pragma unroll
      for (int s = 0; s < 16; ++s) {
        const size_t ko = (size_t)(s < 8 ? s : s + 8) * G4U_;
        o0[s] = Wk[ko + cb0];
        o1[s] = Wk[ko + cb1];
        o2[s] = Wk[ko + cb2];
        o3[s] = Wk[ko + cb3];
      }
    };

    // 2-stage software pipeline on B fragments; A fragments from LDS.
    v16h b0, b1, b2, b3, p0, p1, p2, p3;
    loadB(0, b0, b1, b2, b3);
    #pragma unroll 1
    for (int kb = 0; kb < KTOT_; kb += 32) {
      // A fragment per ISA layout: lanes 0-15 K(0-7,16-23), 16-31 K(8-15,24-31)
      const v8h* pa = (kb < E_)
        ? (const v8h*)(Ax + arow * AXSTR_ + kb + khalf8)
        : (const v8h*)(Ah + arow * AHSTR_ + (kb - E_) + khalf8);
      v8h alo = pa[0];
      v8h ahi = pa[2];   // +16 halfs
      v16h a = __builtin_shufflevector(alo, ahi,
               0,1,2,3,4,5,6,7,8,9,10,11,12,13,14,15);
      const int kn = (kb + 32 < KTOT_) ? (kb + 32) : 0;  // wrap: harmless dup
      loadB(kn, p0, p1, p2, p3);
      z0 = __builtin_amdgcn_wmma_f32_16x16x32_f16(false, a, false, b0, (short)0, z0, false, false);
      z1 = __builtin_amdgcn_wmma_f32_16x16x32_f16(false, a, false, b1, (short)0, z1, false, false);
      z2 = __builtin_amdgcn_wmma_f32_16x16x32_f16(false, a, false, b2, (short)0, z2, false, false);
      z3 = __builtin_amdgcn_wmma_f32_16x16x32_f16(false, a, false, b3, (short)0, z3, false, false);
      b0 = p0; b1 = p1; b2 = p2; b3 = p3;
    }

    // LSTM cell update: fully lane-local, branch-free, c-state in registers
    #pragma unroll
    for (int v = 0; v < 8; ++v) {
      float iv = z0[v], fv = z1[v], gv = z2[v], ov = z3[v];
      float cn = sigmoid_fast(fv) * cst[v] + sigmoid_fast(iv) * tanh_fast(gv);
      float hn = sigmoid_fast(ov) * tanh_fast(cn);
      cst[v] = cn;
      int m = mBase + v;
      Hdir[m * U_ + u] = (_Float16)hn;
      if (t == T_ - 1) Hod[m * U_ + u] = hn;
    }

    // publish h(t) before any WG starts staging step t+1
    grid_barrier(cnt, gen, ++phase);
  }
}

// ---------------- tiny dense head: sigmoid(((hf|hb)@W1+b1)@W2+b2) ----------
__global__ void __launch_bounds__(256) head_dense(
    const float* __restrict__ Hf, const float* __restrict__ Hb,
    const float* __restrict__ W1, const float* __restrict__ b1,
    const float* __restrict__ W2, const float* __restrict__ b2,
    float* __restrict__ out) {
  __shared__ float hid[B_][64];
  int tid = threadIdx.x;
  for (int idx = tid; idx < B_ * 64; idx += 256) {
    int b = idx >> 6, j = idx & 63;
    float s = b1[j];
    const float* hf = Hf + b * U_;
    const float* hb = Hb + b * U_;
    for (int k = 0; k < U_; ++k) s = fmaf(hf[k], W1[k * 64 + j], s);
    for (int k = 0; k < U_; ++k) s = fmaf(hb[k], W1[(U_ + k) * 64 + j], s);
    hid[b][j] = s;
  }
  __syncthreads();
  if (tid < B_) {
    float s = b2[0];
    for (int j = 0; j < 64; ++j) s = fmaf(hid[tid][j], W2[j], s);
    out[tid] = sigmoid_fast(s);
  }
}

// ---------------- host-side launch ----------------
extern "C" void kernel_launch(void* const* d_in, const int* in_sizes, int n_in,
                              void* d_out, int out_size, void* d_ws, size_t ws_size,
                              hipStream_t stream) {
  const int*   sent = (const int*)d_in[0];
  const float* emb  = (const float*)d_in[1];
  const float* Wx_f = (const float*)d_in[2];
  const float* Wh_f = (const float*)d_in[3];
  const float* b_f  = (const float*)d_in[4];
  const float* Wx_b = (const float*)d_in[5];
  const float* Wh_b = (const float*)d_in[6];
  const float* b_b  = (const float*)d_in[7];
  const float* W1   = (const float*)d_in[8];
  const float* b1   = (const float*)d_in[9];
  const float* W2   = (const float*)d_in[10];
  const float* b2   = (const float*)d_in[11];
  float* out = (float*)d_out;
  (void)in_sizes; (void)n_in; (void)out_size; (void)ws_size;

  char* ws = (char*)d_ws;
  size_t off = 0;
  auto take = [&](size_t bytes) -> char* {
    char* p = ws + off;
    off += (bytes + 255) & ~(size_t)255;
    return p;
  };
  unsigned* bar   = (unsigned*)take(256);
  _Float16* X     = (_Float16*)take((size_t)T_ * B_ * E_ * 2);   // 16 MB
  _Float16* Wx16f = (_Float16*)take((size_t)E_ * G4U_ * 2);      // 1 MB
  _Float16* Wh16f = (_Float16*)take((size_t)U_ * G4U_ * 2);      // 2 MB
  _Float16* Wx16b = (_Float16*)take((size_t)E_ * G4U_ * 2);
  _Float16* Wh16b = (_Float16*)take((size_t)U_ * G4U_ * 2);
  _Float16* H16   = (_Float16*)take((size_t)2 * B_ * U_ * 2);
  float*    Hout  = (float*)take((size_t)2 * B_ * U_ * 4);

  init_state<<<64, 256, 0, stream>>>(bar, H16);
  {
    int n4 = E_ * G4U_ / 4;
    convert_f16<<<(n4 + 255) / 256, 256, 0, stream>>>(Wx_f, Wx16f, n4);
    convert_f16<<<(n4 + 255) / 256, 256, 0, stream>>>(Wx_b, Wx16b, n4);
  }
  {
    int n4 = U_ * G4U_ / 4;
    convert_f16<<<(n4 + 255) / 256, 256, 0, stream>>>(Wh_f, Wh16f, n4);
    convert_f16<<<(n4 + 255) / 256, 256, 0, stream>>>(Wh_b, Wh16b, n4);
  }
  {
    int total4 = T_ * B_ * E_ / 4;
    embed_gather_f16<<<(total4 + 255) / 256, 256, 0, stream>>>(sent, emb, X);
  }
  size_t ldsBytes = (size_t)(64 * AXSTR_ + 64 * AHSTR_) * sizeof(_Float16); // 100352 B
  bilstm_persistent<<<2 * NWG_, 256, ldsBytes, stream>>>(
      X, Wx16f, Wh16f, Wx16b, Wh16b, b_f, b_b, H16, Hout, bar);
  head_dense<<<1, 256, 0, stream>>>(Hout, Hout + B_ * U_, W1, b1, W2, b2, out);
}
